// SparseGlobalAttention_71253507441105
// MI455X (gfx1250) — matrix operand order repaired and checked by reading, not compiled, not run on hardware
//
#include <hip/hip_runtime.h>
#include <hip/hip_bf16.h>
#include <math.h>

// ---------------------------------------------------------------------------
// Problem constants (static per the reference)
// ---------------------------------------------------------------------------
constexpr int Bn  = 2;
constexpr int Sn  = 8192;
constexpr int Dn  = 3072;
constexpr int Hn  = 4;
constexpr int HDn = 768;         // head dim
constexpr int NLn = 32;          // landmarks
constexpr long long Mn = (long long)Bn * Sn;   // 16384 rows in the big GEMMs
#define SCALE_F 0.03608439182435161f           // 1/sqrt(768)

typedef __attribute__((ext_vector_type(16))) __bf16 v16bf;
typedef __attribute__((ext_vector_type(8)))  float  v8f;

// landmark index j -> sequence position (numpy linspace(0,511,32) truncation, *16)
__device__ __forceinline__ int lm_index(int j) {
    int sel = (int)((511.0 * (double)j) / 31.0);
    return sel * 16;
}

__device__ __forceinline__ __bf16 f32_to_bf16(float f) {
    unsigned u = __float_as_uint(f);
    unsigned r = (u + 0x7FFFu + ((u >> 16) & 1u)) >> 16;
    unsigned short h = (unsigned short)r;
    __bf16 o;
    __builtin_memcpy(&o, &h, 2);
    return o;
}
__device__ __forceinline__ float bf16_to_f32(__bf16 b) {
    unsigned short h;
    __builtin_memcpy(&h, &b, 2);
    unsigned u = ((unsigned)h) << 16;
    return __uint_as_float(u);
}

// ---------------------------------------------------------------------------
// f32 -> bf16 conversion, 4 elements / thread (n must be multiple of 4)
// ---------------------------------------------------------------------------
__global__ void cvt_bf16_kernel(const float* __restrict__ src,
                                __bf16* __restrict__ dst, long long n4) {
    long long i = (long long)blockIdx.x * blockDim.x + threadIdx.x;
    if (i >= n4) return;
    float4 f = reinterpret_cast<const float4*>(src)[i];
    __bf16 o[4] = {f32_to_bf16(f.x), f32_to_bf16(f.y),
                   f32_to_bf16(f.z), f32_to_bf16(f.w)};
    uint2 u;
    __builtin_memcpy(&u, o, 8);
    reinterpret_cast<uint2*>(dst)[i] = u;
}

// ---------------------------------------------------------------------------
// Transposing f32 -> bf16 weight conversion: Wt[n,k] = bf16(W[k,n])
// 32x32 tiles through LDS; fully coalesced reads and writes.
// ---------------------------------------------------------------------------
__global__ void __launch_bounds__(256)
transpose_cvt_kernel(const float* __restrict__ W, __bf16* __restrict__ Wt) {
    __shared__ float tile[32][33];
    const int tx = threadIdx.x;            // 0..31
    const int ty = threadIdx.y;            // 0..7
    const long long k0 = (long long)blockIdx.x * 32;
    const long long n0 = (long long)blockIdx.y * 32;
#pragma unroll
    for (int i = 0; i < 4; ++i) {
        const int r = ty + i * 8;          // k offset
        tile[r][tx] = W[(k0 + r) * Dn + n0 + tx];
    }
    __syncthreads();
#pragma unroll
    for (int i = 0; i < 4; ++i) {
        const int r = ty + i * 8;          // n offset
        Wt[(n0 + r) * Dn + k0 + tx] = f32_to_bf16(tile[tx][r]);
    }
}

// ---------------------------------------------------------------------------
// landmark_tokens[b,j,:] = hidden[b, li[j], :] + emb[j,:]
// ---------------------------------------------------------------------------
__global__ void lm_build_kernel(const float* __restrict__ hidden,
                                const float* __restrict__ emb,
                                float* __restrict__ lm) {
    long long i = (long long)blockIdx.x * blockDim.x + threadIdx.x;
    if (i >= (long long)Bn * NLn * Dn) return;
    int d = (int)(i % Dn);
    int j = (int)((i / Dn) % NLn);
    int b = (int)(i / ((long long)Dn * NLn));
    int li = lm_index(j);
    lm[i] = hidden[((long long)b * Sn + li) * Dn + d] + emb[(long long)j * Dn + d];
}

// ---------------------------------------------------------------------------
// Small GEMM (64 rows): out[r,n] = bias[n] + sum_k A[r,k] * W[k,n]   (all f32)
// ---------------------------------------------------------------------------
__global__ void small_gemm_kernel(const float* __restrict__ A,
                                  const float* __restrict__ W,
                                  const float* __restrict__ bias,
                                  float* __restrict__ out, int R) {
    long long i = (long long)blockIdx.x * blockDim.x + threadIdx.x;
    if (i >= (long long)R * Dn) return;
    int n = (int)(i % Dn);
    int r = (int)(i / Dn);
    const float* a = A + (long long)r * Dn;
    float acc = bias[n];
    for (int k = 0; k < Dn; ++k) acc = fmaf(a[k], W[(long long)k * Dn + n], acc);
    out[i] = acc;
}

// ---------------------------------------------------------------------------
// WMMA bf16 GEMM: C[M,N] = A[M,K] x Wt[N,K]^T (+bias); optional +addend, f32 out
// Block tile 128x256, 8 waves (2x4), each wave 64x64 = 4x4 WMMA 16x16 tiles,
// BLK_K=32. Both LDS tiles K-contiguous -> every fragment read is ds_load_b128.
// Per wave per K-step: 16 x ds_load_b128 : 16 x v_wmma (1:1).
// ---------------------------------------------------------------------------
#define BLK_M 128
#define BLK_N 256
#define BLK_K 32
#define LDS_S 40    // padded LDS row stride (bf16 elems); 80B = 16B multiple

template <bool FINAL>
__global__ void __launch_bounds__(256)
wmma_gemm_kernel(const __bf16* __restrict__ A, const __bf16* __restrict__ Bt,
                 const float* __restrict__ bias, const float* __restrict__ addend,
                 __bf16* __restrict__ outb, float* __restrict__ outf,
                 int Kdim, int Ndim) {
    __shared__ __attribute__((aligned(16))) __bf16 As[BLK_M * LDS_S];
    __shared__ __attribute__((aligned(16))) __bf16 Bs[BLK_N * LDS_S];

    const int tid  = threadIdx.x;
    const int lane = tid & 31;
    const int wave = tid >> 5;
    const int half = lane >> 4;     // lane half: K/M sub-partitions (ISA 7.12.2)
    const int l16  = lane & 15;
    const int wm   = wave >> 2;     // 0..1  (64 rows each)
    const int wn   = wave & 3;      // 0..3  (64 cols each)
    const long long block_m = (long long)blockIdx.y * BLK_M;
    const long long block_n = (long long)blockIdx.x * BLK_N;

    v8f zero = {};
    v8f acc[4][4];
#pragma unroll
    for (int i = 0; i < 4; ++i)
#pragma unroll
        for (int j = 0; j < 4; ++j) acc[i][j] = zero;

    for (int k0 = 0; k0 < Kdim; k0 += BLK_K) {
        // cooperative tile loads: 16B units; row = u>>2, chunk = (u&3)*8 elems
        // A tile: 128 rows * 4 units = 512 units -> 2 per thread
#pragma unroll
        for (int p = 0; p < 2; ++p) {
            const int u = tid + 256 * p;
            const int r = u >> 2, c = (u & 3) * 8;
            *reinterpret_cast<uint4*>(&As[r * LDS_S + c]) =
                *reinterpret_cast<const uint4*>(A + (block_m + r) * Kdim + k0 + c);
        }
        // B tile: 256 rows * 4 units = 1024 units -> 4 per thread
#pragma unroll
        for (int p = 0; p < 4; ++p) {
            const int u = tid + 256 * p;
            const int r = u >> 2, c = (u & 3) * 8;
            *reinterpret_cast<uint4*>(&Bs[r * LDS_S + c]) =
                *reinterpret_cast<const uint4*>(Bt + (block_n + r) * Kdim + k0 + c);
        }
        if (k0 + BLK_K < Kdim) {  // prefetch next tiles (global_prefetch_b8)
            __builtin_prefetch((const void*)(A + (block_m + (tid >> 1)) * Kdim + k0 + BLK_K), 0, 0);
            __builtin_prefetch((const void*)(Bt + (block_n + tid) * Kdim + k0 + BLK_K), 0, 0);
        }
        __syncthreads();

        // A fragment: elems 0..7 = K half*8.., elems 8..15 = K 16+half*8..
        v16bf afrag[4];
#pragma unroll
        for (int i = 0; i < 4; ++i) {
            const int m0 = wm * 64 + i * 16 + l16;
            uint4 chunks[2];
            chunks[0] = *reinterpret_cast<const uint4*>(&As[m0 * LDS_S + half * 8]);
            chunks[1] = *reinterpret_cast<const uint4*>(&As[m0 * LDS_S + 16 + half * 8]);
            __builtin_memcpy(&afrag[i], chunks, 32);
        }
        // B fragment: lane column n0, elems 0..7 = K half*8.., elems 8..15 = K 16+half*8..
        v16bf bfrag[4];
#pragma unroll
        for (int j = 0; j < 4; ++j) {
            const int n0 = wn * 64 + j * 16 + l16;
            uint4 chunks[2];
            chunks[0] = *reinterpret_cast<const uint4*>(&Bs[n0 * LDS_S + half * 8]);
            chunks[1] = *reinterpret_cast<const uint4*>(&Bs[n0 * LDS_S + 16 + half * 8]);
            __builtin_memcpy(&bfrag[j], chunks, 32);
        }
#pragma unroll
        for (int i = 0; i < 4; ++i)
#pragma unroll
            for (int j = 0; j < 4; ++j)
                acc[i][j] = __builtin_amdgcn_wmma_f32_16x16x32_bf16(
                    false, afrag[i], false, bfrag[j], (short)0, acc[i][j],
                    false, false);
        __syncthreads();
    }

    // epilogue: C/D layout — lane half picks M+8, VGPR v = M row (ISA 7.12.2)
#pragma unroll
    for (int i = 0; i < 4; ++i) {
#pragma unroll
        for (int j = 0; j < 4; ++j) {
            const long long n = block_n + wn * 64 + j * 16 + l16;
            const float bv = bias[n];
#pragma unroll
            for (int v = 0; v < 8; ++v) {
                const long long m = block_m + wm * 64 + i * 16 + half * 8 + v;
                const long long off = m * Ndim + n;
                const float val = acc[i][j][v] + bv;
                if (FINAL) {
                    outf[off] = val + addend[off];
                } else {
                    outb[off] = f32_to_bf16(val);
                }
            }
        }
    }
}

// ---------------------------------------------------------------------------
// Global attention: every token attends to 32 landmarks. One block per token.
// ---------------------------------------------------------------------------
__global__ void __launch_bounds__(128)
global_attn_kernel(const __bf16* __restrict__ Qb, const float* __restrict__ Klm,
                   const float* __restrict__ Vlm, __bf16* __restrict__ ctx) {
    const int token = blockIdx.x;          // 0..B*S-1
    const int b = token / Sn;
    const int t = threadIdx.x;             // 128 threads
    const int h = t >> 5, j = t & 31;

    __shared__ float sc[Hn * NLn];
    __shared__ float w[Hn * NLn];
    __shared__ float red[Hn * 2];

    // scores: vectorized x8 (q: uint4 of 8 bf16; k: 2x float4)
    const uint4*  qv = reinterpret_cast<const uint4*>(
        Qb + (long long)token * Dn + h * HDn);
    const float4* kv = reinterpret_cast<const float4*>(
        Klm + ((long long)b * NLn + j) * Dn + h * HDn);
    float acc = 0.f;
    for (int x = 0; x < HDn / 8; ++x) {
        uint4 qq = qv[x];
        __bf16 qa[8];
        __builtin_memcpy(qa, &qq, 16);
        float4 k0 = kv[2 * x], k1 = kv[2 * x + 1];
        acc = fmaf(bf16_to_f32(qa[0]), k0.x, acc);
        acc = fmaf(bf16_to_f32(qa[1]), k0.y, acc);
        acc = fmaf(bf16_to_f32(qa[2]), k0.z, acc);
        acc = fmaf(bf16_to_f32(qa[3]), k0.w, acc);
        acc = fmaf(bf16_to_f32(qa[4]), k1.x, acc);
        acc = fmaf(bf16_to_f32(qa[5]), k1.y, acc);
        acc = fmaf(bf16_to_f32(qa[6]), k1.z, acc);
        acc = fmaf(bf16_to_f32(qa[7]), k1.w, acc);
    }
    sc[t] = acc * SCALE_F;
    __syncthreads();
    if (j == 0) {
        float m = sc[h * NLn];
        for (int x = 1; x < NLn; ++x) m = fmaxf(m, sc[h * NLn + x]);
        float s = 0.f;
        for (int x = 0; x < NLn; ++x) s += __expf(sc[h * NLn + x] - m);
        red[h * 2] = m;
        red[h * 2 + 1] = s;
    }
    __syncthreads();
    w[t] = __expf(sc[t] - red[h * 2]) / red[h * 2 + 1];
    __syncthreads();
    for (int i = 0; i < Dn / 128; ++i) {
        const int d  = t + i * 128;
        const int hh = d / HDn;
        float a = 0.f;
        for (int x = 0; x < NLn; ++x)
            a = fmaf(w[hh * NLn + x], Vlm[((long long)b * NLn + x) * Dn + d], a);
        ctx[(long long)token * Dn + d] = f32_to_bf16(a);
    }
}

// ---------------------------------------------------------------------------
// Landmark attention: scores over all S tokens, softmax, weighted V sum
// ---------------------------------------------------------------------------
__global__ void lm_scores_kernel(const float* __restrict__ Qlm,
                                 const __bf16* __restrict__ Kb,
                                 float* __restrict__ scores) {
    long long i = (long long)blockIdx.x * blockDim.x + threadIdx.x;
    if (i >= (long long)Bn * Hn * NLn * Sn) return;
    int s = (int)(i % Sn);
    int j = (int)((i / Sn) % NLn);
    int h = (int)((i / ((long long)Sn * NLn)) % Hn);
    int b = (int)(i / ((long long)Sn * NLn * Hn));
    const float4* qv = reinterpret_cast<const float4*>(
        Qlm + ((long long)b * NLn + j) * Dn + h * HDn);
    const uint4*  kv = reinterpret_cast<const uint4*>(
        Kb + ((long long)b * Sn + s) * Dn + h * HDn);
    float acc = 0.f;
    for (int x = 0; x < HDn / 8; ++x) {
        uint4 kk = kv[x];
        __bf16 ka[8];
        __builtin_memcpy(ka, &kk, 16);
        float4 q0 = qv[2 * x], q1 = qv[2 * x + 1];
        acc = fmaf(q0.x, bf16_to_f32(ka[0]), acc);
        acc = fmaf(q0.y, bf16_to_f32(ka[1]), acc);
        acc = fmaf(q0.z, bf16_to_f32(ka[2]), acc);
        acc = fmaf(q0.w, bf16_to_f32(ka[3]), acc);
        acc = fmaf(q1.x, bf16_to_f32(ka[4]), acc);
        acc = fmaf(q1.y, bf16_to_f32(ka[5]), acc);
        acc = fmaf(q1.z, bf16_to_f32(ka[6]), acc);
        acc = fmaf(q1.w, bf16_to_f32(ka[7]), acc);
    }
    scores[i] = acc * SCALE_F;
}

__global__ void __launch_bounds__(256)
lm_softmax_kernel(float* __restrict__ scores) {
    __shared__ float red[256];
    float* row = scores + (long long)blockIdx.x * Sn;
    const int t = threadIdx.x;
    float m = -INFINITY;
    for (int s = t; s < Sn; s += 256) m = fmaxf(m, row[s]);
    red[t] = m;
    __syncthreads();
    for (int o = 128; o; o >>= 1) {
        if (t < o) red[t] = fmaxf(red[t], red[t + o]);
        __syncthreads();
    }
    m = red[0];
    __syncthreads();
    float sum = 0.f;
    for (int s = t; s < Sn; s += 256) sum += __expf(row[s] - m);
    red[t] = sum;
    __syncthreads();
    for (int o = 128; o; o >>= 1) {
        if (t < o) red[t] += red[t + o];
        __syncthreads();
    }
    const float inv = 1.f / red[0];
    for (int s = t; s < Sn; s += 256) row[s] = __expf(row[s] - m) * inv;
}

__global__ void __launch_bounds__(256)
lm_ctx_kernel(const float* __restrict__ w, const __bf16* __restrict__ Vb,
              float* __restrict__ ctx) {
    const int chunks = HDn / 256;               // 3
    int blk = blockIdx.x;
    int c = blk % chunks;
    int j = (blk / chunks) % NLn;
    int h = (blk / (chunks * NLn)) % Hn;
    int b = blk / (chunks * NLn * Hn);
    const int d = c * 256 + threadIdx.x;        // within head
    const float*  wr = w + (((long long)b * Hn + h) * NLn + j) * Sn;
    const __bf16* v  = Vb + h * HDn + d;
    float acc = 0.f;
    for (int s = 0; s < Sn; ++s)
        acc = fmaf(wr[s], bf16_to_f32(v[((long long)b * Sn + s) * Dn]), acc);
    ctx[((long long)b * NLn + j) * Dn + h * HDn + d] = acc;
}

// landmark rows of d_out: currently global_out + hidden; fix to global_out + lm_out
__global__ void lm_scatter_kernel(const float* __restrict__ lm_out,
                                  const float* __restrict__ hidden,
                                  float* __restrict__ out) {
    long long i = (long long)blockIdx.x * blockDim.x + threadIdx.x;
    if (i >= (long long)Bn * NLn * Dn) return;
    int d = (int)(i % Dn);
    int j = (int)((i / Dn) % NLn);
    int b = (int)(i / ((long long)Dn * NLn));
    const long long m = ((long long)b * Sn + lm_index(j)) * Dn + d;
    out[m] += lm_out[i] - hidden[m];
}

// ---------------------------------------------------------------------------
// Launch
// ---------------------------------------------------------------------------
extern "C" void kernel_launch(void* const* d_in, const int* in_sizes, int n_in,
                              void* d_out, int out_size, void* d_ws, size_t ws_size,
                              hipStream_t stream) {
    (void)in_sizes; (void)n_in; (void)out_size; (void)ws_size;
    const float* hidden = (const float*)d_in[0];
    const float* emb    = (const float*)d_in[1];
    const float* Wq = (const float*)d_in[2]; const float* bq = (const float*)d_in[3];
    const float* Wk = (const float*)d_in[4]; const float* bk = (const float*)d_in[5];
    const float* Wv = (const float*)d_in[6]; const float* bv = (const float*)d_in[7];
    const float* Wo = (const float*)d_in[8]; const float* bo = (const float*)d_in[9];
    float* out = (float*)d_out;

    char* ws = (char*)d_ws;
    size_t off = 0;
    auto alloc = [&](size_t bytes) -> char* {
        char* p = ws + off;
        off = (off + bytes + 255) & ~(size_t)255;
        return p;
    };
    const size_t MD  = (size_t)Mn * Dn;
    const size_t DD  = (size_t)Dn * Dn;
    const size_t LMD = (size_t)Bn * NLn * Dn;

    __bf16* hid_b  = (__bf16*)alloc(MD * 2);
    __bf16* WqT_b  = (__bf16*)alloc(DD * 2);   // transposed bf16 weights [n][k]
    __bf16* WkT_b  = (__bf16*)alloc(DD * 2);
    __bf16* WvT_b  = (__bf16*)alloc(DD * 2);
    __bf16* WoT_b  = (__bf16*)alloc(DD * 2);
    __bf16* Q_b    = (__bf16*)alloc(MD * 2);
    __bf16* K_b    = (__bf16*)alloc(MD * 2);
    __bf16* V_b    = (__bf16*)alloc(MD * 2);
    __bf16* ctxg_b = (__bf16*)alloc(MD * 2);
    float* lm_tok  = (float*)alloc(LMD * 4);
    float* Qlm     = (float*)alloc(LMD * 4);
    float* Klm     = (float*)alloc(LMD * 4);
    float* Vlm     = (float*)alloc(LMD * 4);
    float* scores  = (float*)alloc((size_t)Bn * Hn * NLn * Sn * 4);
    float* ctx_lm  = (float*)alloc(LMD * 4);
    float* lm_out  = (float*)alloc(LMD * 4);

    auto blocks = [](long long n, int t) { return (unsigned)((n + t - 1) / t); };

    // 1. downcast activations; transpose+downcast weights
    cvt_bf16_kernel<<<blocks(MD / 4, 256), 256, 0, stream>>>(hidden, hid_b, (long long)(MD / 4));
    dim3 tgrid(Dn / 32, Dn / 32), tblk(32, 8);
    transpose_cvt_kernel<<<tgrid, tblk, 0, stream>>>(Wq, WqT_b);
    transpose_cvt_kernel<<<tgrid, tblk, 0, stream>>>(Wk, WkT_b);
    transpose_cvt_kernel<<<tgrid, tblk, 0, stream>>>(Wv, WvT_b);
    transpose_cvt_kernel<<<tgrid, tblk, 0, stream>>>(Wo, WoT_b);

    // 2. landmark tokens + small projections (64 rows, f32 VALU)
    lm_build_kernel<<<blocks(LMD, 256), 256, 0, stream>>>(hidden, emb, lm_tok);
    const int R = Bn * NLn;
    small_gemm_kernel<<<blocks(LMD, 256), 256, 0, stream>>>(lm_tok, Wq, bq, Qlm, R);
    small_gemm_kernel<<<blocks(LMD, 256), 256, 0, stream>>>(lm_tok, Wk, bk, Klm, R);
    small_gemm_kernel<<<blocks(LMD, 256), 256, 0, stream>>>(lm_tok, Wv, bv, Vlm, R);

    // 3. big WMMA GEMMs: Q/K/V for all tokens (bf16 out)
    dim3 ggrid(Dn / BLK_N, (unsigned)(Mn / BLK_M));
    wmma_gemm_kernel<false><<<ggrid, 256, 0, stream>>>(hid_b, WqT_b, bq, nullptr, Q_b, nullptr, Dn, Dn);
    wmma_gemm_kernel<false><<<ggrid, 256, 0, stream>>>(hid_b, WkT_b, bk, nullptr, K_b, nullptr, Dn, Dn);
    wmma_gemm_kernel<false><<<ggrid, 256, 0, stream>>>(hid_b, WvT_b, bv, nullptr, V_b, nullptr, Dn, Dn);

    // 4. global attention (tokens -> landmarks), then out-proj fused with +hidden
    global_attn_kernel<<<(unsigned)Mn, 128, 0, stream>>>(Q_b, Klm, Vlm, ctxg_b);
    wmma_gemm_kernel<true><<<ggrid, 256, 0, stream>>>(ctxg_b, WoT_b, bo, hidden, nullptr, out, Dn, Dn);

    // 5. landmark attention (landmarks -> all tokens)
    lm_scores_kernel<<<blocks((long long)Bn * Hn * NLn * Sn, 256), 256, 0, stream>>>(Qlm, K_b, scores);
    lm_softmax_kernel<<<Bn * Hn * NLn, 256, 0, stream>>>(scores);
    lm_ctx_kernel<<<Bn * Hn * NLn * (HDn / 256), 256, 0, stream>>>(scores, V_b, ctx_lm);
    small_gemm_kernel<<<blocks(LMD, 256), 256, 0, stream>>>(ctx_lm, Wo, bo, lm_out, R);

    // 6. replace landmark rows: out = global_out + lm_out
    lm_scatter_kernel<<<blocks(LMD, 256), 256, 0, stream>>>(lm_out, hidden, out);
}
